// CausalSelfAttention_74148315398674
// MI455X (gfx1250) — compile-verified, matrix operand order checked by reading, not run on hardware
//
#include <hip/hip_runtime.h>
#include <hip/hip_bf16.h>

typedef __bf16 bf16;
typedef bf16  v16bf __attribute__((ext_vector_type(16)));
typedef bf16  v8bf  __attribute__((ext_vector_type(8)));
typedef float v8f   __attribute__((ext_vector_type(8)));

static constexpr int B_  = 4;
static constexpr int N_  = 2048;
static constexpr int D_  = 1024;
static constexpr int H_  = 16;
static constexpr int HD_ = 64;   // head dim
static constexpr int E_  = 3 * HD_;  // 192 proj channels (k,q,v)

// ---------------------------------------------------------------------------
// D = A(16xK) * B(Kx16) + C with bf16 inputs, f32 accumulate
static __device__ __forceinline__ v8f wmma_bf(v16bf a, v16bf b, v8f c) {
  return __builtin_amdgcn_wmma_f32_16x16x32_bf16(false, a, false, b, (short)0, c,
                                                 false, false);
}

// Load one 16x32 bf16 WMMA fragment (A-layout; identical pattern serves as the
// B-layout for the transposed operand). lane (l&15) owns row row16+(l&15);
// low half-wave reads K {kbase..kbase+7, kbase+16..23}, high half-wave reads
// K {kbase+8..15, kbase+24..31}. Rows must be 16B aligned at kbase.
static __device__ __forceinline__ v16bf frag_ld(const bf16* mat, int stride,
                                                int row16, int kbase) {
  const int lane = threadIdx.x & 31;
  const bf16* p = mat + (size_t)(row16 + (lane & 15)) * stride + kbase +
                  ((lane >> 4) << 3);
  v8bf lo = *(const v8bf*)p;
  v8bf hi = *(const v8bf*)(p + 16);
  v16bf r;
#pragma unroll
  for (int i = 0; i < 8; ++i) { r[i] = lo[i]; r[i + 8] = hi[i]; }
  return r;
}

// ---------------------------------------------------------------------------
// elementwise f32 -> bf16
__global__ __launch_bounds__(256) void cvt_f32_bf16(const float* __restrict__ in,
                                                    bf16* __restrict__ out,
                                                    long n) {
  long i = (long)blockIdx.x * blockDim.x + threadIdx.x;
  if (i < n) out[i] = (bf16)in[i];
}

// Wkqv [H][HD][E] f32  ->  WkqvT [H][E][HD] bf16
__global__ __launch_bounds__(256) void cvt_wkqvT(const float* __restrict__ w,
                                                 bf16* __restrict__ wt) {
  int i = blockIdx.x * blockDim.x + threadIdx.x;
  if (i >= H_ * E_ * HD_) return;
  int d = i & (HD_ - 1);
  int e = (i >> 6) % E_;
  int h = i / (E_ * HD_);
  wt[i] = (bf16)w[((size_t)h * HD_ + d) * E_ + e];
}

// ---------------------------------------------------------------------------
// QKV projection: proj[b,n,h,e] = sum_d x[b,n,h*64+d]*Wkqv[h,d,e] + bkqv[h,e]
// K -> Km [BH][N][HD], Q -> Qm [BH][N][HD], V -> Vt [BH][HD][N] (transposed)
__global__ __launch_bounds__(256) void qkv_kernel(const bf16* __restrict__ xbf,
                                                  const bf16* __restrict__ wkT,
                                                  const float* __restrict__ bkqv,
                                                  bf16* __restrict__ Qm,
                                                  bf16* __restrict__ Km,
                                                  bf16* __restrict__ Vt) {
  const int lane = threadIdx.x & 31;
  const int wave = threadIdx.x >> 5;
  const int bh = blockIdx.y;              // b*H + h
  const int b = bh >> 4, h = bh & (H_ - 1);
  const int t0 = blockIdx.x * 128 + wave * 16;   // token tile base

  const bf16* xmat = xbf + (size_t)b * N_ * D_ + h * HD_;  // rows stride D_
  const bf16* wmat = wkT + (size_t)h * E_ * HD_;           // rows stride HD_
  const float* bias = bkqv + h * E_;

  // token fragments over the 64-wide contraction (2 chunks of 32)
  v16bf xf0 = frag_ld(xmat, D_, t0, 0);
  v16bf xf1 = frag_ld(xmat, D_, t0, 32);

  const int hi8 = (lane >> 4) << 3;

  // --- K and Q tiles: swapped orientation (M=channels, N=tokens) ---
#pragma unroll
  for (int et = 0; et < 8; ++et) {
    const int e0 = et * 16;
    v16bf wf0 = frag_ld(wmat, HD_, e0, 0);
    v16bf wf1 = frag_ld(wmat, HD_, e0, 32);
    v8f c = {};
    c = wmma_bf(wf0, xf0, c);
    c = wmma_bf(wf1, xf1, c);
    // c[r] = proj[t0+(lane&15)][e0 + hi8 + r]
    const int tok = t0 + (lane & 15);
    const int eb = e0 + hi8;
    v8bf pack;
#pragma unroll
    for (int r = 0; r < 8; ++r) pack[r] = (bf16)(c[r] + bias[eb + r]);
    bf16* dst = (et < 4)
                    ? (Km + ((size_t)bh * N_ + tok) * HD_ + eb)         // e<64: K
                    : (Qm + ((size_t)bh * N_ + tok) * HD_ + (eb - 64)); // q
    *(v8bf*)dst = pack;
  }

  // --- V tiles: normal orientation (M=tokens, N=channels), store transposed ---
#pragma unroll
  for (int et = 8; et < 12; ++et) {
    const int e0 = et * 16;
    v16bf wf0 = frag_ld(wmat, HD_, e0, 0);
    v16bf wf1 = frag_ld(wmat, HD_, e0, 32);
    v8f c = {};
    c = wmma_bf(xf0, wf0, c);
    c = wmma_bf(xf1, wf1, c);
    // c[r] = proj[t0 + hi8 + r][e0 + (lane&15)]
    const int d = (e0 - 128) + (lane & 15);
    const float bv = bias[e0 + (lane & 15)];
    const int tokb = t0 + hi8;
    v8bf pack;
#pragma unroll
    for (int r = 0; r < 8; ++r) pack[r] = (bf16)(c[r] + bv);
    *(v8bf*)(Vt + ((size_t)bh * HD_ + d) * N_ + tokb) = pack;
  }
}

// ---------------------------------------------------------------------------
// Flash attention, causal. One wave = 16 query rows. sa [B][N][D] bf16.
__global__ __launch_bounds__(256) void attn_kernel(const bf16* __restrict__ Qm,
                                                   const bf16* __restrict__ Km,
                                                   const bf16* __restrict__ Vt,
                                                   bf16* __restrict__ sa) {
  const int lane = threadIdx.x & 31;
  const int wave = threadIdx.x >> 5;
  const int bh = blockIdx.y;
  const int b = bh >> 4, h = bh & (H_ - 1);
  const int q0 = blockIdx.x * 128 + wave * 16;

  const bf16* Qp = Qm + (size_t)bh * N_ * HD_;
  const bf16* Kp = Km + (size_t)bh * N_ * HD_;
  const bf16* Vp = Vt + (size_t)bh * HD_ * N_;

  __shared__ __align__(16) bf16 plds[8][16 * 40];  // P transpose staging, padded
  bf16* myp = &plds[wave][0];

  v16bf qa0 = frag_ld(Qp, HD_, q0, 0);
  v16bf qa1 = frag_ld(Qp, HD_, q0, 32);

  v8f o[4];
#pragma unroll
  for (int c = 0; c < 4; ++c) o[c] = (v8f){};
  float mrun[8], lrun[8];
#pragma unroll
  for (int r = 0; r < 8; ++r) { mrun[r] = -3.0e38f; lrun[r] = 0.0f; }

  const int hi8 = (lane >> 4) << 3;
  const int ncol = lane & 15;
  const int irow0 = q0 + hi8;  // query row of accumulator slot 0
  const int kend = q0 + 16;

  for (int kb = 0; kb < kend; kb += 32) {
    // scores: two 16(query)x16(key) tiles, contraction over HD=64
    v8f s[2];
#pragma unroll
    for (int t = 0; t < 2; ++t) {
      v16bf kf0 = frag_ld(Kp, HD_, kb + 16 * t, 0);
      v16bf kf1 = frag_ld(Kp, HD_, kb + 16 * t, 32);
      v8f c = {};
      c = wmma_bf(qa0, kf0, c);
      c = wmma_bf(qa1, kf1, c);
      s[t] = c;
    }
    // scale + causal mask
#pragma unroll
    for (int t = 0; t < 2; ++t)
#pragma unroll
      for (int r = 0; r < 8; ++r) {
        const int j = kb + 16 * t + ncol;
        const int i = irow0 + r;
        s[t][r] = (j <= i) ? s[t][r] * 0.125f : -3.0e38f;
      }
    // online softmax per row (row spans 16 lanes of a half-wave)
    float alpha[8];
#pragma unroll
    for (int r = 0; r < 8; ++r) {
      float mx = fmaxf(s[0][r], s[1][r]);
#pragma unroll
      for (int off = 8; off >= 1; off >>= 1) mx = fmaxf(mx, __shfl_xor(mx, off, 16));
      const float mnew = fmaxf(mrun[r], mx);
      const float p0 = __expf(s[0][r] - mnew);
      const float p1 = __expf(s[1][r] - mnew);
      s[0][r] = p0;
      s[1][r] = p1;
      float ps = p0 + p1;
#pragma unroll
      for (int off = 8; off >= 1; off >>= 1) ps += __shfl_xor(ps, off, 16);
      alpha[r] = __expf(mrun[r] - mnew);
      lrun[r] = lrun[r] * alpha[r] + ps;
      mrun[r] = mnew;
    }
#pragma unroll
    for (int c = 0; c < 4; ++c)
#pragma unroll
      for (int r = 0; r < 8; ++r) o[c][r] *= alpha[r];

    // C-layout P -> A-layout via wave-private LDS tile (16 rows, stride 40)
#pragma unroll
    for (int t = 0; t < 2; ++t)
#pragma unroll
      for (int r = 0; r < 8; ++r)
        myp[(hi8 + r) * 40 + 16 * t + ncol] = (bf16)s[t][r];
    asm volatile("s_wait_dscnt 0" ::: "memory");
    v16bf pf = frag_ld(myp, 40, 0, 0);

    // O += P (16x32) * V (32 x HD), V fragments contiguous from transposed Vt
#pragma unroll
    for (int c = 0; c < 4; ++c) {
      v16bf vf = frag_ld(Vp, N_, 16 * c, kb);
      o[c] = wmma_bf(pf, vf, o[c]);
    }
  }

  // normalize and scatter into sa [B][N][H*HD]
#pragma unroll
  for (int r = 0; r < 8; ++r) {
    const float inv = 1.0f / lrun[r];
    const int row = irow0 + r;
#pragma unroll
    for (int c = 0; c < 4; ++c) {
      const int col = h * HD_ + 16 * c + ncol;
      sa[((size_t)b * N_ + row) * D_ + col] = (bf16)(o[c][r] * inv);
    }
  }
}

// ---------------------------------------------------------------------------
// out[t][o] = sum_d sa[t][d] * Wout[o][d] + bout[o]   (swapped orientation)
__global__ __launch_bounds__(256) void outproj_kernel(const bf16* __restrict__ sa,
                                                      const bf16* __restrict__ wbf,
                                                      const float* __restrict__ bout,
                                                      float* __restrict__ out) {
  const int lane = threadIdx.x & 31;
  const int wave = threadIdx.x >> 5;
  const int t0 = (blockIdx.x * 8 + wave) * 16;  // token tile
  const int o0 = blockIdx.y * 64;               // channel tile (64 wide)

  v8f acc[4];
#pragma unroll
  for (int c = 0; c < 4; ++c) acc[c] = (v8f){};

  for (int kc = 0; kc < D_; kc += 32) {
    v16bf sfrag = frag_ld(sa, D_, t0, kc);  // B operand: token rows
#pragma unroll
    for (int c = 0; c < 4; ++c) {
      v16bf wfrag = frag_ld(wbf, D_, o0 + 16 * c, kc);  // A operand: channel rows
      acc[c] = wmma_bf(wfrag, sfrag, acc[c]);
    }
  }

  const int tok = t0 + (lane & 15);
  const int hi8 = (lane >> 4) << 3;
#pragma unroll
  for (int c = 0; c < 4; ++c) {
    const int ob = o0 + 16 * c + hi8;  // 8 consecutive channels in slots
    float* dst = out + (size_t)tok * D_ + ob;
    float4 v0 = make_float4(acc[c][0] + bout[ob + 0], acc[c][1] + bout[ob + 1],
                            acc[c][2] + bout[ob + 2], acc[c][3] + bout[ob + 3]);
    float4 v1 = make_float4(acc[c][4] + bout[ob + 4], acc[c][5] + bout[ob + 5],
                            acc[c][6] + bout[ob + 6], acc[c][7] + bout[ob + 7]);
    *(float4*)dst = v0;
    *(float4*)(dst + 4) = v1;
  }
}

// ---------------------------------------------------------------------------
extern "C" void kernel_launch(void* const* d_in, const int* in_sizes, int n_in,
                              void* d_out, int out_size, void* d_ws, size_t ws_size,
                              hipStream_t stream) {
  const float* x    = (const float*)d_in[0];  // [B,N,D]
  const float* Wkqv = (const float*)d_in[1];  // [H,HD,3HD]
  const float* bkqv = (const float*)d_in[2];  // [H,3HD]
  const float* Wout = (const float*)d_in[3];  // [D,D]
  const float* bout = (const float*)d_in[4];  // [D]
  float* out = (float*)d_out;                 // [B,N,D]

  size_t off = 0;
  auto carve = [&](size_t bytes) {
    void* p = (char*)d_ws + off;
    off += (bytes + 255) & ~(size_t)255;
    return p;
  };
  bf16* xbf   = (bf16*)carve((size_t)B_ * N_ * D_ * 2);        // 16 MB
  bf16* wkT   = (bf16*)carve((size_t)H_ * E_ * HD_ * 2);       // 384 KB
  bf16* wobf  = (bf16*)carve((size_t)D_ * D_ * 2);             // 2 MB
  bf16* Qm    = (bf16*)carve((size_t)B_ * H_ * N_ * HD_ * 2);  // 16 MB
  bf16* Km    = (bf16*)carve((size_t)B_ * H_ * N_ * HD_ * 2);  // 16 MB
  bf16* Vt    = (bf16*)carve((size_t)B_ * H_ * HD_ * N_ * 2);  // 16 MB
  bf16* sa    = xbf;  // xbf is dead after qkv_kernel; reuse for attention output

  {
    long n = (long)B_ * N_ * D_;
    cvt_f32_bf16<<<dim3((n + 255) / 256), dim3(256), 0, stream>>>(x, xbf, n);
  }
  {
    long n = (long)D_ * D_;
    cvt_f32_bf16<<<dim3((n + 255) / 256), dim3(256), 0, stream>>>(Wout, wobf, n);
  }
  {
    int n = H_ * E_ * HD_;
    cvt_wkqvT<<<dim3((n + 255) / 256), dim3(256), 0, stream>>>(Wkqv, wkT);
  }

  qkv_kernel<<<dim3(N_ / 128, B_ * H_), dim3(256), 0, stream>>>(xbf, wkT, bkqv,
                                                                Qm, Km, Vt);
  attn_kernel<<<dim3(N_ / 128, B_ * H_), dim3(256), 0, stream>>>(Qm, Km, Vt, sa);
  outproj_kernel<<<dim3(B_ * N_ / 128, D_ / 64), dim3(256), 0, stream>>>(sa, wobf,
                                                                         bout, out);
  (void)in_sizes; (void)n_in; (void)out_size; (void)ws_size;
}
